// ESAN_17514876634157
// MI455X (gfx1250) — matrix operand order repaired and checked by reading, not compiled, not run on hardware
//
#include <hip/hip_runtime.h>
#include <hip/hip_bf16.h>

typedef __attribute__((ext_vector_type(16))) _Float16 v16h;
typedef __attribute__((ext_vector_type(8)))  _Float16 v8h;
typedef __attribute__((ext_vector_type(8)))  float    v8f;

namespace {
constexpr int S      = 64;
constexpr int NSUB   = 2048;
constexpr int E      = 32768;
constexpr int IN_DIM = 128;
constexpr int HID    = 256;
constexpr int OUT    = 64;
constexpr int NNODES = 100000;
constexpr int KMAX   = 256;     // largest K over the three GEMMs
constexpr int KPAD   = 8;       // halves of padding per LDS column (bank skew)
constexpr int NT     = 2;       // 16-wide N tiles per block (32 columns)
}

// ---------------- utility: zero a float buffer (grid-stride) ----------------
__global__ void k_zero(float* __restrict__ p, long long n) {
  long long i = blockIdx.x * (long long)blockDim.x + threadIdx.x;
  long long stride = (long long)gridDim.x * blockDim.x;
  for (; i < n; i += stride) p[i] = 0.0f;
}

// ---------------- degree scatter (in-degree over dst) -----------------------
__global__ void k_deg(const int* __restrict__ ei, float* __restrict__ deg) {
  int idx = blockIdx.x * blockDim.x + threadIdx.x;
  if (idx >= S * E) return;
  int g = idx / E, e = idx - g * E;
  int dst = ei[(size_t)g * 2 * E + E + e];
  atomicAdd(&deg[g * NSUB + dst], 1.0f);
}

// deg -> rsqrt(deg + 1)   (self-loop adds 1, so deg >= 1 always)
__global__ void k_norm(float* __restrict__ deg) {
  int i = blockIdx.x * blockDim.x + threadIdx.x;
  if (i < S * NSUB) deg[i] = rsqrtf(deg[i] + 1.0f);
}

// ---------------- WMMA GEMM: Out[g] = X[g] @ W  (+bias) (*norm[row]) --------
// Block = 256 threads (8 waves). Each block owns a 32-wide N panel; the W
// panel (K x 32) is converted to f16 once and staged in LDS *column-major*
// (stride K+KPAD halves) so a lane's B fragment -- 16 consecutive K values of
// one column, in this code's contiguous k order -- is two contiguous ds_load_b128.
// Each wave computes a 64x32 strip: per K-step it builds 2 B fragments (LDS)
// and 4 A fragments (global b128 + cvt_pk), issuing 8 v_wmma. A-matrix
// re-reads drop from N/16 to N/32 block sweeps.
//   A (16x32 f16): h=lane>>4, row=lane&15; elem t -> K=k0+16h+t
//   B (32x16 f16): col=lane&15; elem t -> K=k0+16h+t
//   D (16x16 f32): col=lane&15; elem v -> row=8h+v
template <bool HAS_NORM, bool HAS_BIAS>
__global__ __launch_bounds__(256)
void k_gemm_wmma(const float* __restrict__ X, const float* __restrict__ W,
                 const float* __restrict__ norm, const float* __restrict__ bias,
                 float* __restrict__ Out, int M, int K, int N)
{
  __shared__ __align__(32) _Float16 sW[16 * NT * (KMAX + KPAD)];

  const int g    = blockIdx.z;
  const int tid  = threadIdx.x;
  const int lane = tid & 31;
  const int wave = tid >> 5;
  const int col0 = blockIdx.x * (16 * NT);    // first column of this block
  const int half = lane >> 4;
  const int r    = lane & 15;
  const int ldc  = K + KPAD;                  // LDS column stride (halves)

  // stage W panel (columns col0 .. col0+31, all K) into LDS, col-major f16
  for (int i = tid; i < K * 16 * NT; i += 256) {
    const int c = i & (16 * NT - 1), k = i >> 5;
    sW[c * ldc + k] = (_Float16)W[(size_t)k * N + col0 + c];
  }
  __syncthreads();

  const float* Xg = X + (size_t)g * M * K;
  float*       Og = Out + (size_t)g * M * N;

  const int tm0 = (blockIdx.y * 8 + wave) * 4;   // first of 4 M-tiles

  v8f acc[4][NT] = {};
  for (int k0 = 0; k0 < K; k0 += 32) {
    // B fragments: two contiguous 16B LDS loads per N tile from lane's column
    v16h b[NT];
#pragma unroll
    for (int nt = 0; nt < NT; ++nt) {
      const _Float16* bp = &sW[(nt * 16 + r) * ldc + k0 + half * 16];
      const v8h blo = *(const v8h*)(bp);
      const v8h bhi = *(const v8h*)(bp + 8);
      b[nt] = __builtin_shufflevector(blo, bhi, 0, 1, 2, 3, 4, 5, 6, 7,
                                      8, 9, 10, 11, 12, 13, 14, 15);
    }
#pragma unroll
    for (int mt = 0; mt < 4; ++mt) {
      const float* ap = Xg + (size_t)((tm0 + mt) * 16 + r) * K + k0 + half * 16;
      if (mt == 0 && k0 + 32 < K) __builtin_prefetch(ap + 32, 0, 0);
      v16h a;
#pragma unroll
      for (int t = 0; t < 8; ++t) {
        a[t]     = (_Float16)ap[t];
        a[8 + t] = (_Float16)ap[8 + t];
      }
#pragma unroll
      for (int nt = 0; nt < NT; ++nt) {
        acc[mt][nt] = __builtin_amdgcn_wmma_f32_16x16x32_f16(
            false, a, false, b[nt], (short)0, acc[mt][nt], false, false);
      }
    }
  }

#pragma unroll
  for (int nt = 0; nt < NT; ++nt) {
    const int col = col0 + nt * 16 + r;
    const float bcol = HAS_BIAS ? bias[col] : 0.0f;
#pragma unroll
    for (int mt = 0; mt < 4; ++mt) {
#pragma unroll
      for (int v = 0; v < 8; ++v) {
        const int row = (tm0 + mt) * 16 + half * 8 + v;
        float val = acc[mt][nt][v] + bcol;
        if (HAS_NORM) val *= norm[(size_t)g * M + row];
        Og[(size_t)row * N + col] = val;
      }
    }
  }
}

// ---------------- per-edge message scatter: agg[dst] += Hs[src] -------------
// thread = (g, e, 4-channel chunk); float4 gather + 4x f32 atomics (L2-resident)
__global__ void k_edge_scatter(const int* __restrict__ ei,
                               const float* __restrict__ Hs,
                               float* __restrict__ agg)
{
  int idx  = blockIdx.x * blockDim.x + threadIdx.x;     // S*E*(HID/4)
  int c4   = idx & (HID / 4 - 1);                       // 0..63
  int rest = idx >> 6;
  int e    = rest & (E - 1);                            // E = 2^15
  int g    = rest >> 15;
  if (g >= S) return;
  int src = ei[(size_t)g * 2 * E + e];
  int dst = ei[(size_t)g * 2 * E + E + e];
  const float4 m = *(const float4*)(Hs + ((size_t)g * NSUB + src) * HID + c4 * 4);
  float* o = agg + ((size_t)g * NSUB + dst) * HID + c4 * 4;
  atomicAdd(o + 0, m.x);
  atomicAdd(o + 1, m.y);
  atomicAdd(o + 2, m.z);
  atomicAdd(o + 3, m.w);
}

// h = (relu?) (h * norm[node] + bias[c])
__global__ void k_finalize(float* __restrict__ h, const float* __restrict__ norm,
                           const float* __restrict__ bias, int do_relu)
{
  int idx = blockIdx.x * blockDim.x + threadIdx.x;      // S*NSUB*HID
  if (idx >= S * NSUB * HID) return;
  int c  = idx & (HID - 1);
  int gn = idx >> 8;                                    // HID = 256
  float v = h[idx] * norm[gn] + bias[c];
  if (do_relu) v = fmaxf(v, 0.0f);
  h[idx] = v;
}

// ---------------- global node scatter-mean ----------------------------------
__global__ void k_counts(const int* __restrict__ nid, float* __restrict__ counts) {
  int i = blockIdx.x * blockDim.x + threadIdx.x;
  if (i >= S * NSUB) return;
  atomicAdd(&counts[nid[i]], 1.0f);
}

__global__ void k_node_scatter(const float* __restrict__ P,
                               const int* __restrict__ nid,
                               float* __restrict__ acc)
{
  int idx = blockIdx.x * blockDim.x + threadIdx.x;      // S*NSUB*(OUT/4)
  if (idx >= S * NSUB * (OUT / 4)) return;
  int c4 = idx & 15;
  int gn = idx >> 4;
  int id = nid[gn];
  const float4 p = *(const float4*)(P + (size_t)gn * OUT + c4 * 4);
  float* o = acc + (size_t)id * OUT + c4 * 4;
  atomicAdd(o + 0, p.x);
  atomicAdd(o + 1, p.y);
  atomicAdd(o + 2, p.z);
  atomicAdd(o + 3, p.w);
}

__global__ void k_logsoftmax(const float* __restrict__ acc,
                             const float* __restrict__ counts,
                             float* __restrict__ out)
{
  int n = blockIdx.x * blockDim.x + threadIdx.x;
  if (n >= NNODES) return;
  float inv = 1.0f / fmaxf(counts[n], 1.0f);
  const float* a = acc + (size_t)n * OUT;
  float m = -3.402823466e38f;
  for (int c = 0; c < OUT; ++c) m = fmaxf(m, a[c] * inv);
  float s = 0.0f;
  for (int c = 0; c < OUT; ++c) s += expf(a[c] * inv - m);
  const float ls = logf(s);
  float* o = out + (size_t)n * OUT;
  for (int c = 0; c < OUT; ++c) o[c] = a[c] * inv - m - ls;
}

// ---------------- pipeline ---------------------------------------------------
extern "C" void kernel_launch(void* const* d_in, const int* in_sizes, int n_in,
                              void* d_out, int out_size, void* d_ws, size_t ws_size,
                              hipStream_t stream)
{
  (void)in_sizes; (void)n_in; (void)out_size; (void)ws_size;

  const float* x  = (const float*)d_in[0];
  const float* W1 = (const float*)d_in[1];
  const float* b1 = (const float*)d_in[2];
  const float* W2 = (const float*)d_in[3];
  const float* b2 = (const float*)d_in[4];
  const float* Wa = (const float*)d_in[5];
  const float* ba = (const float*)d_in[6];
  const int*   ei  = (const int*)d_in[7];
  const int*   nid = (const int*)d_in[8];
  float* out = (float*)d_out;

  char* ws = (char*)d_ws;
  const size_t HB       = (size_t)S * NSUB * HID * sizeof(float);   // 128 MB
  const size_t NORM_OFF = 0;
  const size_t A_OFF    = 512ull * 1024;
  const size_t B_OFF    = A_OFF + HB;
  const size_t NODE_OFF = B_OFF + HB;
  const size_t CNT_OFF  = NODE_OFF + (size_t)NNODES * OUT * sizeof(float);

  float* normb = (float*)(ws + NORM_OFF);   // deg, then rsqrt in place
  float* bufA  = (float*)(ws + A_OFF);      // GEMM outputs / P
  float* bufB  = (float*)(ws + B_OFF);      // aggregation / H1 / H2
  float* nacc  = (float*)(ws + NODE_OFF);   // [NNODES, OUT]
  float* cnts  = (float*)(ws + CNT_OFF);    // [NNODES]  (adjacent to nacc)

  // degrees -> norm
  k_zero<<<512, 256, 0, stream>>>(normb, (long long)S * NSUB);
  k_deg<<<(S * E) / 256, 256, 0, stream>>>(ei, normb);
  k_norm<<<(S * NSUB) / 256, 256, 0, stream>>>(normb);

  // conv1: Hs = (X @ W1) * norm[row]
  k_gemm_wmma<true, false><<<dim3(HID / 32, NSUB / 16 / 32, S), 256, 0, stream>>>(
      x, W1, normb, nullptr, bufA, NSUB, IN_DIM, HID);
  hipMemcpyAsync(bufB, bufA, HB, hipMemcpyDeviceToDevice, stream);      // self-loops
  k_edge_scatter<<<(S * E * (HID / 4)) / 256, 256, 0, stream>>>(ei, bufA, bufB);
  k_finalize<<<(S * NSUB * HID) / 256, 256, 0, stream>>>(bufB, normb, b1, 1);

  // conv2: Hs2 = (H1 @ W2) * norm[row]
  k_gemm_wmma<true, false><<<dim3(HID / 32, NSUB / 16 / 32, S), 256, 0, stream>>>(
      bufB, W2, normb, nullptr, bufA, NSUB, HID, HID);
  hipMemcpyAsync(bufB, bufA, HB, hipMemcpyDeviceToDevice, stream);      // self-loops
  k_edge_scatter<<<(S * E * (HID / 4)) / 256, 256, 0, stream>>>(ei, bufA, bufB);
  k_finalize<<<(S * NSUB * HID) / 256, 256, 0, stream>>>(bufB, normb, b2, 0);

  // head: P = H2 @ Wa + ba   (into bufA, only needs 32 MB of it)
  k_gemm_wmma<false, true><<<dim3(OUT / 32, NSUB / 16 / 32, S), 256, 0, stream>>>(
      bufB, Wa, nullptr, ba, bufA, NSUB, HID, OUT);

  // scatter-mean over node ids, then log-softmax
  k_zero<<<4096, 256, 0, stream>>>(nacc, (long long)NNODES * OUT + NNODES);
  k_counts<<<(S * NSUB) / 256, 256, 0, stream>>>(nid, cnts);
  k_node_scatter<<<(S * NSUB * (OUT / 4)) / 256, 256, 0, stream>>>(bufA, nid, nacc);
  k_logsoftmax<<<(NNODES + 255) / 256, 256, 0, stream>>>(nacc, cnts, out);
}
